// MultiheadA_19439021981875
// MI455X (gfx1250) — matrix operand order repaired and checked by reading, not compiled, not run on hardware
//
#include <hip/hip_runtime.h>
#include <hip/hip_bf16.h>

// ---------------------------------------------------------------------------
// MultiheadA on MI455X (gfx1250): bf16 WMMA everywhere, f32 accumulation.
//   B=4, S=2048, D=1024, H=16, HD=64.  Compute-bound (~137 GFLOP vs ~100MB).
// Pipeline:
//   1) convert weights fp32->bf16 (Wq|Wk|Wv concatenated as [3072,1024], Wo)
//   2) transpose-convert x -> xT bf16 [D, B*S]
//   3) fused QKV GEMM (v_wmma_f32_16x16x32_bf16) + bias + mel-rotary epilogue
//   4) flash attention: one block per (b,h, 256-query slab); 8 waves share
//      K/V tiles staged in LDS via double-buffered global_load_async_to_lds
//      (ASYNCcnt/s_wait_asynccnt), online softmax, P transposed via per-wave
//      LDS bounce, O written as AT [D, B*S]
//   5) output GEMM + bias -> d_out fp32 [B,S,D]
// ---------------------------------------------------------------------------

typedef __bf16 bf16;
typedef __attribute__((ext_vector_type(16))) __bf16 v16bf;
typedef __attribute__((ext_vector_type(8)))  float  v8f;

#define SQ    2048
#define DIM   1024
#define NTOK  8192          // B*S
#define HEADS 16
#define HD    64

#define WMMA_BF16(a, b, c) \
  __builtin_amdgcn_wmma_f32_16x16x32_bf16(false, (a), false, (b), (short)0, (c), false, false)

union Frag { v16bf v; uint4 u[2]; };

// A-fragment (16x32, 16-bit): lane = M row; elems 0..7 = K[8*hi .. 8*hi+7],
// elems 8..15 = K[16+8*hi .. 23+8*hi].  p points at row start + kblk + 8*hi.
__device__ __forceinline__ v16bf ldA(const bf16* p) {
  Frag f;
  f.u[0] = *(const uint4*)(p);
  f.u[1] = *(const uint4*)(p + 16);
  return f.v;
}
// B-fragment (32x16, 16-bit): lane = N column; elems 0..7 = K[8*hi .. 8*hi+7],
// elems 8..15 = K[16+8*hi .. 23+8*hi].  p points at row (kblk + 8*hi), col n0 + lr.
__device__ __forceinline__ v16bf ldB(const bf16* p, int ld) {
  v16bf v;
#pragma unroll
  for (int s = 0; s < 8; ++s) {
    v[s]     = p[(size_t)s * ld];
    v[8 + s] = p[(size_t)(16 + s) * ld];
  }
  return v;
}
__device__ __forceinline__ v8f zero8() {
  v8f z = {0.f, 0.f, 0.f, 0.f, 0.f, 0.f, 0.f, 0.f};
  return z;
}

// CDNA5 async copy: 16B global -> LDS, tracked by ASYNCcnt.
// lds_off = low 32 bits of the generic LDS address.
__device__ __forceinline__ void async_cp16(unsigned lds_off, const void* gptr) {
  asm volatile("global_load_async_to_lds_b128 %0, %1, off"
               :: "v"(lds_off), "v"(gptr) : "memory");
}
__device__ __forceinline__ void wait_async0() {
  asm volatile("s_wait_asynccnt 0x0" ::: "memory");
}

// ---------------------------------------------------------------------------
// fp32 -> bf16 elementwise convert (n multiple of 4)
// ---------------------------------------------------------------------------
__global__ void __launch_bounds__(256)
cvt_bf16_kernel(const float* __restrict__ src, bf16* __restrict__ dst, int n) {
  int i = (blockIdx.x * blockDim.x + threadIdx.x) * 4;
  if (i < n) {
    float4 f = *(const float4*)(src + i);
    bf16 o[4] = {(bf16)f.x, (bf16)f.y, (bf16)f.z, (bf16)f.w};
    *(uint2*)(dst + i) = *(uint2*)o;
  }
}

// ---------------------------------------------------------------------------
// x [rows, cols] fp32  ->  xT [cols, rows] bf16   (LDS tile transpose)
// ---------------------------------------------------------------------------
__global__ void __launch_bounds__(256)
transpose_cvt_kernel(const float* __restrict__ x, bf16* __restrict__ xT,
                     int rows, int cols) {
  __shared__ float tile[32][33];
  const int c0 = blockIdx.x * 32, r0 = blockIdx.y * 32;
  const int tx = threadIdx.x, ty = threadIdx.y;   // block (32, 8)
#pragma unroll
  for (int i = ty; i < 32; i += 8)
    tile[i][tx] = x[(size_t)(r0 + i) * cols + c0 + tx];
  __syncthreads();
#pragma unroll
  for (int i = ty; i < 32; i += 8)
    xT[(size_t)(c0 + i) * rows + r0 + tx] = (bf16)tile[tx][i];
}

// ---------------------------------------------------------------------------
// Fused QKV GEMM + bias + rotary.
//   C[m, n] = sum_k Wqkv[m, k] * xT[k, n]        M=3072 (Q|K|V), N=8192, K=1024
//   one wave = 64x64 tile (4x4 wmma tiles), 8 waves/block, grid = 6144/8.
//   C layout: elem r of lane l -> M = r + 8*(l/16), N = l%16  => rotary pairs
//   (even feature, odd feature) are adjacent accumulator elements per lane.
// ---------------------------------------------------------------------------
__global__ void __launch_bounds__(256)
qkv_gemm_kernel(const bf16* __restrict__ W, const bf16* __restrict__ xT,
                const float* __restrict__ qb, const float* __restrict__ vb,
                const float* __restrict__ theta_p,
                bf16* __restrict__ Qr, bf16* __restrict__ KT,
                bf16* __restrict__ Vm) {
  const int lane = threadIdx.x & 31, hi = lane >> 4, lr = lane & 15;
  const int wid = blockIdx.x * 8 + (threadIdx.x >> 5);
  const int m0 = (wid >> 7) * 64;          // 48 m-tiles
  const int n0 = (wid & 127) * 64;         // 128 n-tiles

  v8f acc[4][4];
#pragma unroll
  for (int i = 0; i < 4; ++i)
#pragma unroll
    for (int j = 0; j < 4; ++j) acc[i][j] = zero8();

  for (int k = 0; k < DIM; k += 32) {
    if (k + 32 < DIM) {
      __builtin_prefetch(W + (size_t)(m0 + lr) * DIM + k + 32, 0, 3);
      __builtin_prefetch(xT + (size_t)(k + 32 + lane) * NTOK + n0, 0, 3);
    }
    v16bf a[4], b[4];
#pragma unroll
    for (int mi = 0; mi < 4; ++mi)
      a[mi] = ldA(W + (size_t)(m0 + mi * 16 + lr) * DIM + k + 8 * hi);
#pragma unroll
    for (int ni = 0; ni < 4; ++ni)
      b[ni] = ldB(xT + (size_t)(k + 8 * hi) * NTOK + n0 + ni * 16 + lr, NTOK);
#pragma unroll
    for (int mi = 0; mi < 4; ++mi)
#pragma unroll
      for (int ni = 0; ni < 4; ++ni)
        acc[mi][ni] = WMMA_BF16(a[mi], b[ni], acc[mi][ni]);
  }

  const float theta   = *theta_p;
  const float melstep = 2595.0f * log10f(1.0f + 8000.0f / 700.0f) / 31.0f;

#pragma unroll
  for (int mi = 0; mi < 4; ++mi) {
    const int mbase = m0 + mi * 16;
    const int which = mbase >> 10;         // 0=Q, 1=K, 2=V
    const int mm    = mbase & 1023;
    const int h     = mm >> 6;
    const int hd0   = mm & 63;             // 16-tile within a head (64 = 4 tiles)
#pragma unroll
    for (int ni = 0; ni < 4; ++ni) {
      const int tok  = n0 + ni * 16 + lr;
      const int bidx = tok >> 11;
      const int spos = tok & 2047;
      if (which == 2) {                    // ---- V: bias, row-major [B,H,S,HD]
#pragma unroll
        for (int r = 0; r < 8; ++r) {
          const int mo = r + 8 * hi;
          float val = acc[mi][ni][r] + vb[mm + mo];
          Vm[(((size_t)(bidx * HEADS + h)) * SQ + spos) * HD + hd0 + mo] = (bf16)val;
        }
      } else {                             // ---- Q/K: rotary on (even, odd) pairs
#pragma unroll
        for (int r = 0; r < 8; r += 2) {
          const int mo = r + 8 * hi;       // even
          float va = acc[mi][ni][r];
          float vo = acc[mi][ni][r + 1];
          if (which == 0) { va += qb[mm + mo]; vo += qb[mm + mo + 1]; }
          const int   hd   = hd0 + mo;
          const float lin  = (float)(hd >> 1) * melstep;
          const float freq = theta * (700.0f / 220000.0f) *
                             (exp10f(lin * (1.0f / 2595.0f)) - 1.0f);
          float sn, cs;
          __sincosf((float)spos * freq, &sn, &cs);
          float o0 = va * cs - vo * sn;
          float o1 = va * sn + vo * cs;
          if (which == 0) {                // fold softmax scale 1/sqrt(64) into Q
            o0 *= 0.125f; o1 *= 0.125f;
            const size_t base =
                (((size_t)(bidx * HEADS + h)) * SQ + spos) * HD + hd;
            Qr[base] = (bf16)o0; Qr[base + 1] = (bf16)o1;
          } else {                         // K stored transposed [B,H,HD,S]
            const size_t base =
                (((size_t)(bidx * HEADS + h)) * HD + hd) * SQ + spos;
            KT[base] = (bf16)o0; KT[base + SQ] = (bf16)o1;
          }
        }
      }
    }
  }
}

// ---------------------------------------------------------------------------
// Flash attention.  One block per (b*h, 256-query slab): 64 bh * 8 = 512 blocks.
// All 8 waves share the same (b,h): K tile [64 hd][32 kpos] and V tile
// [32 kpos][64 hd] are staged in LDS with double-buffered
// global_load_async_to_lds_b128 (1 coalesced 16B chunk per thread, ASYNCcnt),
// cutting K/V L2 traffic 8x vs per-wave loads.
//   scores C[q, kpos]: A = Q row-major, B = Ksh (k-major)  -> 2 wmma per tile
//   P (C layout) -> per-wave LDS bounce -> A layout, then O += P @ V.
// ---------------------------------------------------------------------------
__global__ void __launch_bounds__(256)
attn_kernel(const bf16* __restrict__ Qr, const bf16* __restrict__ KT,
            const bf16* __restrict__ Vm, bf16* __restrict__ AT) {
  __shared__ __align__(16) bf16 Ksh[2][64][32];     // [buf][hd][kpos]   4KB each
  __shared__ __align__(16) bf16 Vsh[2][32][64];     // [buf][kpos][hd]   4KB each
  __shared__ __align__(16) bf16 Pt[8][2][16][32];   // [wave][qsub][q][kpos] 16KB

  const int tid  = threadIdx.x;
  const int lane = tid & 31, hi = lane >> 4, lr = lane & 15;
  const int wv   = tid >> 5;
  const int bh   = blockIdx.x >> 3;                 // 0..63
  const int q0   = ((blockIdx.x & 7) * 8 + wv) * 32;

  const bf16* Qb  = Qr + (size_t)bh * SQ * HD;
  const bf16* KTb = KT + (size_t)bh * HD * SQ;
  const bf16* Vb  = Vm + (size_t)bh * SQ * HD;

  // this thread's fixed staging assignment (one 16B chunk per tile each)
  const int krow = tid >> 2, kcol = (tid & 3) * 8;  // K: 64 rows x 4 chunks
  const int vrow = tid >> 3, vcol = (tid & 7) * 8;  // V: 32 rows x 8 chunks

  v16bf qa[2][2];
#pragma unroll
  for (int s = 0; s < 2; ++s)
#pragma unroll
    for (int c = 0; c < 2; ++c)
      qa[s][c] = ldA(Qb + (size_t)(q0 + s * 16 + lr) * HD + c * 32 + 8 * hi);

  v8f   o[2][4];
  float mrow[2][8], lrow[2][8];
#pragma unroll
  for (int s = 0; s < 2; ++s) {
#pragma unroll
    for (int t = 0; t < 4; ++t) o[s][t] = zero8();
#pragma unroll
    for (int r = 0; r < 8; ++r) { mrow[s][r] = -1e30f; lrow[s][r] = 0.f; }
  }

  // prologue: stage tile 0 into buffer 0
  async_cp16((unsigned)(size_t)&Ksh[0][krow][kcol], KTb + (size_t)krow * SQ + kcol);
  async_cp16((unsigned)(size_t)&Vsh[0][vrow][vcol], Vb + (size_t)vrow * HD + vcol);
  wait_async0();
  __syncthreads();

  for (int it = 0; it < SQ / 32; ++it) {
    const int kp  = it * 32;
    const int buf = it & 1;
    if (it + 1 < SQ / 32) {                // overlap next tile's async copies
      const int kp1 = kp + 32;
      async_cp16((unsigned)(size_t)&Ksh[buf ^ 1][krow][kcol],
                 KTb + (size_t)krow * SQ + kp1 + kcol);
      async_cp16((unsigned)(size_t)&Vsh[buf ^ 1][vrow][vcol],
                 Vb + (size_t)(kp1 + vrow) * HD + vcol);
    }
    // ---- scores: 8 wmma (2 qsub x 2 kpos-subtiles x 2 hd-halves), B from LDS
    v16bf kb[2][2];
#pragma unroll
    for (int kt = 0; kt < 2; ++kt)
#pragma unroll
      for (int c = 0; c < 2; ++c)
        kb[kt][c] = ldB(&Ksh[buf][c * 32 + 8 * hi][kt * 16 + lr], 32);
    v8f sa[2][2];
#pragma unroll
    for (int s = 0; s < 2; ++s)
#pragma unroll
      for (int kt = 0; kt < 2; ++kt) {
        v8f t = zero8();
        t = WMMA_BF16(qa[s][0], kb[kt][0], t);
        t = WMMA_BF16(qa[s][1], kb[kt][1], t);
        sa[s][kt] = t;
      }
    // ---- online softmax per q-row (row spans the 16 lanes of this half)
#pragma unroll
    for (int s = 0; s < 2; ++s) {
#pragma unroll
      for (int r = 0; r < 8; ++r) {
        float mx = fmaxf(sa[s][0][r], sa[s][1][r]);
#pragma unroll
        for (int d = 1; d < 16; d <<= 1) mx = fmaxf(mx, __shfl_xor(mx, d, 32));
        const float mn = fmaxf(mrow[s][r], mx);
        const float sc = __expf(mrow[s][r] - mn);
        mrow[s][r] = mn;
        const float p0 = __expf(sa[s][0][r] - mn);
        const float p1 = __expf(sa[s][1][r] - mn);
        float sum = p0 + p1;
#pragma unroll
        for (int d = 1; d < 16; d <<= 1) sum += __shfl_xor(sum, d, 32);
        lrow[s][r] = lrow[s][r] * sc + sum;
        Pt[wv][s][r + 8 * hi][lr]      = (bf16)p0;
        Pt[wv][s][r + 8 * hi][16 + lr] = (bf16)p1;
#pragma unroll
        for (int t = 0; t < 4; ++t) o[s][t][r] *= sc;
      }
    }
    asm volatile("" ::: "memory");   // keep per-wave LDS store->load order
    // ---- O += P @ V : 8 wmma (2 qsub x 4 hd tiles), K = 32 key positions
    v16bf vbf[4];
#pragma unroll
    for (int t = 0; t < 4; ++t)
      vbf[t] = ldB(&Vsh[buf][8 * hi][t * 16 + lr], 64);
#pragma unroll
    for (int s = 0; s < 2; ++s) {
      Frag pf;
      pf.u[0] = *(const uint4*)&Pt[wv][s][lr][8 * hi];
      pf.u[1] = *(const uint4*)&Pt[wv][s][lr][16 + 8 * hi];
#pragma unroll
      for (int t = 0; t < 4; ++t)
        o[s][t] = WMMA_BF16(pf.v, vbf[t], o[s][t]);
    }
    // next tile's async copies must land before anyone flips buffers
    wait_async0();
    __syncthreads();
  }

  // ---- epilogue: normalize and write transposed AT[D, B*S] for the o-proj GEMM
  const int b = bh >> 4, h = bh & 15;
#pragma unroll
  for (int s = 0; s < 2; ++s)
#pragma unroll
    for (int t = 0; t < 4; ++t)
#pragma unroll
      for (int r = 0; r < 8; ++r) {
        const int q  = q0 + s * 16 + r + 8 * hi;
        const int hd = t * 16 + lr;
        AT[(size_t)(h * HD + hd) * NTOK + b * SQ + q] =
            (bf16)(o[s][t][r] / lrow[s][r]);
      }
}

// ---------------------------------------------------------------------------
// Output projection: out[n, m] = sum_k Wo[m, k] * AT[k, n] + ob[m]   (fp32 out)
// ---------------------------------------------------------------------------
__global__ void __launch_bounds__(256)
oproj_gemm_kernel(const bf16* __restrict__ Wo, const bf16* __restrict__ AT,
                  const float* __restrict__ ob, float* __restrict__ out) {
  const int lane = threadIdx.x & 31, hi = lane >> 4, lr = lane & 15;
  const int wid = blockIdx.x * 8 + (threadIdx.x >> 5);
  const int m0 = (wid >> 7) * 64;          // 16 m-tiles
  const int n0 = (wid & 127) * 64;         // 128 n-tiles

  v8f acc[4][4];
#pragma unroll
  for (int i = 0; i < 4; ++i)
#pragma unroll
    for (int j = 0; j < 4; ++j) acc[i][j] = zero8();

  for (int k = 0; k < DIM; k += 32) {
    if (k + 32 < DIM) {
      __builtin_prefetch(Wo + (size_t)(m0 + lr) * DIM + k + 32, 0, 3);
      __builtin_prefetch(AT + (size_t)(k + 32 + lane) * NTOK + n0, 0, 3);
    }
    v16bf a[4], b[4];
#pragma unroll
    for (int mi = 0; mi < 4; ++mi)
      a[mi] = ldA(Wo + (size_t)(m0 + mi * 16 + lr) * DIM + k + 8 * hi);
#pragma unroll
    for (int ni = 0; ni < 4; ++ni)
      b[ni] = ldB(AT + (size_t)(k + 8 * hi) * NTOK + n0 + ni * 16 + lr, NTOK);
#pragma unroll
    for (int mi = 0; mi < 4; ++mi)
#pragma unroll
      for (int ni = 0; ni < 4; ++ni)
        acc[mi][ni] = WMMA_BF16(a[mi], b[ni], acc[mi][ni]);
  }

#pragma unroll
  for (int mi = 0; mi < 4; ++mi)
#pragma unroll
    for (int ni = 0; ni < 4; ++ni) {
      const int tok = n0 + ni * 16 + lr;
#pragma unroll
      for (int r = 0; r < 8; ++r) {
        const int m = m0 + mi * 16 + r + 8 * hi;
        out[(size_t)tok * DIM + m] = acc[mi][ni][r] + ob[m];
      }
    }
}

// ---------------------------------------------------------------------------
extern "C" void kernel_launch(void* const* d_in, const int* in_sizes, int n_in,
                              void* d_out, int out_size, void* d_ws, size_t ws_size,
                              hipStream_t stream) {
  (void)in_sizes; (void)n_in; (void)out_size; (void)ws_size;
  const float* x     = (const float*)d_in[0];
  const float* q_w   = (const float*)d_in[1];
  const float* q_b   = (const float*)d_in[2];
  const float* k_w   = (const float*)d_in[3];
  const float* v_w   = (const float*)d_in[4];
  const float* v_b   = (const float*)d_in[5];
  const float* o_w   = (const float*)d_in[6];
  const float* o_b   = (const float*)d_in[7];
  const float* theta = (const float*)d_in[8];

  // workspace layout (bytes); total 72 MB (xT region reused for AT)
  char* ws = (char*)d_ws;
  const size_t MB = (size_t)1 << 20;
  bf16* xT   = (bf16*)(ws + 0 * MB);      // 16 MB [1024, 8192], later reused as AT
  bf16* Wqkv = (bf16*)(ws + 16 * MB);     //  6 MB [3072, 1024]
  bf16* Wo   = (bf16*)(ws + 22 * MB);     //  2 MB [1024, 1024]
  bf16* Qr   = (bf16*)(ws + 24 * MB);     // 16 MB [B,H,S,HD]
  bf16* KT   = (bf16*)(ws + 40 * MB);     // 16 MB [B,H,HD,S]
  bf16* Vm   = (bf16*)(ws + 56 * MB);     // 16 MB [B,H,S,HD]
  bf16* AT   = xT;                        // 16 MB [1024, 8192]

  const int wn = DIM * DIM;
  cvt_bf16_kernel<<<wn / 1024, 256, 0, stream>>>(q_w, Wqkv,          wn);
  cvt_bf16_kernel<<<wn / 1024, 256, 0, stream>>>(k_w, Wqkv + wn,     wn);
  cvt_bf16_kernel<<<wn / 1024, 256, 0, stream>>>(v_w, Wqkv + 2 * wn, wn);
  cvt_bf16_kernel<<<wn / 1024, 256, 0, stream>>>(o_w, Wo,            wn);

  transpose_cvt_kernel<<<dim3(DIM / 32, NTOK / 32), dim3(32, 8), 0, stream>>>(
      x, xT, NTOK, DIM);

  // 3072/64 * 8192/64 = 6144 wave tiles / 8 waves per block
  qkv_gemm_kernel<<<768, 256, 0, stream>>>(Wqkv, xT, q_b, v_b, theta, Qr, KT, Vm);

  // 64 (b*h) x 8 query-slabs = 512 blocks, 8 waves each
  attn_kernel<<<512, 256, 0, stream>>>(Qr, KT, Vm, AT);

  // 1024/64 * 8192/64 = 2048 wave tiles / 8
  oproj_gemm_kernel<<<256, 256, 0, stream>>>(Wo, AT, o_b, (float*)d_out);
}
